// BasicalBlock_31190052503718
// MI455X (gfx1250) — matrix operand order repaired and checked by reading, not compiled, not run on hardware
//
#include <hip/hip_runtime.h>
#include <hip/hip_bf16.h>

#define BSZ 2
#define CCH 96
#define HH 96
#define WW 96
#define LL (HH*WW)            // 9216
#define DI 192
#define MTOT (BSZ*LL)         // 18432
#define NSTATE 16
#define RRANK 6
#define NPAD 48               // DT_RANK + 2*D_STATE = 38, padded to 48 for 16-wide N tiles
#define NCHUNK 64             // parallel-scan chunks
#define CHLEN (LL / NCHUNK)   // 144 steps per chunk

typedef __attribute__((ext_vector_type(16))) __bf16 v16bf;
typedef __attribute__((ext_vector_type(8)))  float  v8f;

struct B32 { uint4 lo, hi; };   // 32 bytes of bf16 (16 consecutive K values)

// ---------------- weight staging: fp32 -> bf16 ----------------
__global__ void cvt_bf16_kernel(__bf16* __restrict__ dst, const float* __restrict__ src, int n) {
    int i = blockIdx.x * blockDim.x + threadIdx.x;
    if (i < n) dst[i] = (__bf16)src[i];
}

// x_proj_w (4,38,192) -> (4,48,192) zero-padded bf16
__global__ void cvt_xproj_kernel(__bf16* __restrict__ dst, const float* __restrict__ src) {
    int i = blockIdx.x * blockDim.x + threadIdx.x;
    if (i >= 4 * NPAD * DI) return;
    int c = i % DI; int r = (i / DI) % NPAD; int k = i / (DI * NPAD);
    float v = (r < RRANK + 2 * NSTATE) ? src[((size_t)(k * (RRANK + 2 * NSTATE) + r)) * DI + c] : 0.0f;
    dst[i] = (__bf16)v;
}

// ---------------- LayerNorm over channels of NCHW input -> bf16 rows ----------------
__global__ void ln_in_kernel(const float* __restrict__ x, const float* __restrict__ g,
                             const float* __restrict__ be, __bf16* __restrict__ out) {
    int idx = blockIdx.x * blockDim.x + threadIdx.x;
    if (idx >= MTOT) return;
    int b = idx / LL, pos = idx % LL;
    const float* xp = x + (size_t)b * CCH * LL + pos;
    float mean = 0.f;
    for (int c = 0; c < CCH; ++c) mean += xp[(size_t)c * LL];
    mean *= (1.0f / CCH);
    float var = 0.f;
    for (int c = 0; c < CCH; ++c) { float d = xp[(size_t)c * LL] - mean; var += d * d; }
    var *= (1.0f / CCH);
    float inv = rsqrtf(var + 1e-6f);
    __bf16* op = out + (size_t)idx * CCH;
    for (int c = 0; c < CCH; ++c) op[c] = (__bf16)((xp[(size_t)c * LL] - mean) * inv * g[c] + be[c]);
}

// ---------------- depthwise 3x3 conv + SiLU; also init y accumulator with u*sum_k(Ds) --------
__global__ void conv_silu_kernel(const float* __restrict__ xz, const float* __restrict__ cw,
                                 const float* __restrict__ cb, const float* __restrict__ Ds,
                                 __bf16* __restrict__ xiact, float* __restrict__ yacc) {
    int i = blockIdx.x * blockDim.x + threadIdx.x;
    if (i >= MTOT * DI) return;
    int d = i % DI; int p = i / DI;
    int b = p / LL; int pos = p % LL;
    int h = pos / WW, w = pos % WW;
    float acc = cb[d];
#pragma unroll
    for (int dy = -1; dy <= 1; ++dy) {
        int hh = h + dy; if (hh < 0 || hh >= HH) continue;
#pragma unroll
        for (int dx = -1; dx <= 1; ++dx) {
            int ww2 = w + dx; if (ww2 < 0 || ww2 >= WW) continue;
            acc += xz[((size_t)(b * LL + hh * WW + ww2)) * (2 * DI) + d] * cw[d * 9 + (dy + 1) * 3 + (dx + 1)];
        }
    }
    float s = acc / (1.0f + __expf(-acc));      // SiLU
    xiact[(size_t)p * DI + d] = (__bf16)s;
    float sd = Ds[d] + Ds[DI + d] + Ds[2 * DI + d] + Ds[3 * DI + d];
    yacc[(size_t)p * DI + d] = s * sd;          // y + u*Ds term, summed over 4 directions
}

// ---------------- WMMA bf16 GEMM with LDS-staged weights + pipelined A loads --------------
// Weight slab for this block (NT*16 rows x Ktot, ldw == Ktot) is copied to LDS once, shared
// by all 4 waves; A fragments are loaded one k-step ahead so WMMA overlaps global loads.
// MODE 0: plain fp32 store; blockIdx.z offsets W (wzs) and out (ozs)
// MODE 1: leaky-relu, bf16 store
// MODE 2: += resid_NCHW[b,n,pos]*scale[n], fp32 store (row-major, ldc=96)
// MODE 3: += resid[m,n]*scale[n], store fp32 to NCHW layout (final output)
template<int MODE, int NT>
__global__ __launch_bounds__(128)
void gemm_wmma_kernel(const __bf16* __restrict__ A, int lda,
                      const __bf16* __restrict__ Wb, long long wzs,
                      const float* __restrict__ bias,
                      void* __restrict__ outp, int ldc, long long ozs,
                      int Ktot,
                      const float* __restrict__ resid,
                      const float* __restrict__ scale)
{
    __shared__ __align__(16) __bf16 sW[4 * 16 * 192];   // up to 24 KB weight slab
    const int tid = threadIdx.x;
    const int wave = tid >> 5;
    const int lane = tid & 31;
    const int m0 = (blockIdx.x * 4 + wave) * 16;
    const int n0 = blockIdx.y * (NT * 16);
    const int mrow = lane & 15;
    const int hi16 = (lane >> 4) & 1;

    // cooperative stage of the block's weight rows [n0, n0+NT*16) into LDS
    {
        const uint4* wsrc = (const uint4*)(Wb + (size_t)blockIdx.z * wzs + (size_t)n0 * Ktot);
        uint4* wdst = (uint4*)sW;
        const int chunks = NT * 16 * Ktot / 8;
        for (int i = tid; i < chunks; i += 128) wdst[i] = wsrc[i];
    }
    __syncthreads();

    v8f acc[NT];
#pragma unroll
    for (int t = 0; t < NT; ++t) acc[t] = (v8f){0.f, 0.f, 0.f, 0.f, 0.f, 0.f, 0.f, 0.f};

    const __bf16* Arow = A + (size_t)(m0 + mrow) * lda;
    // A fragment loader (16x32 bf16: lane holds 16 consecutive K, K+16 for hi lanes)
    auto load_a = [&](int k0) {
        v16bf af;
#pragma unroll
        for (int v = 0; v < 8; ++v) {
            int kk = (hi16 << 4) | (v << 1);
            union { unsigned u; __bf16 h2[2]; } cv;
            cv.u = *(const unsigned*)(Arow + k0 + kk);
            af[2 * v] = cv.h2[0];
            af[2 * v + 1] = cv.h2[1];
        }
        return af;
    };

    v16bf afc = load_a(0);
    for (int k0 = 0; k0 < Ktot; k0 += 32) {
        v16bf afn = afc;
        if (k0 + 32 < Ktot) afn = load_a(k0 + 32);   // prefetch next A fragment
#pragma unroll
        for (int t = 0; t < NT; ++t) {
            // B fragment from LDS: lane n holds 16 consecutive K (K+16 for hi lanes)
            const __bf16* wp = sW + (size_t)(t * 16 + mrow) * Ktot + k0 + (hi16 << 4);
            B32 raw = *(const B32*)wp;
            v16bf bf = __builtin_bit_cast(v16bf, raw);
            acc[t] = __builtin_amdgcn_wmma_f32_16x16x32_bf16(false, afc, false, bf,
                                                             (short)0, acc[t], false, false);
        }
        afc = afn;
    }

#pragma unroll
    for (int t = 0; t < NT; ++t) {
        int n = n0 + t * 16 + mrow;
        float bv = bias ? bias[n] : 0.0f;
#pragma unroll
        for (int r = 0; r < 8; ++r) {
            int m = m0 + r + hi16 * 8;       // C/D layout: VGPR r -> row r (+8 for hi lanes)
            float v = acc[t][r] + bv;
            if (MODE == 0) {
                float* o = (float*)outp + (size_t)blockIdx.z * ozs;
                o[(size_t)m * ldc + n] = v;
            } else if (MODE == 1) {
                v = (v >= 0.f) ? v : 0.01f * v;
                ((__bf16*)outp)[(size_t)m * ldc + n] = (__bf16)v;
            } else if (MODE == 2) {
                int b = m / LL, pos = m % LL;
                v += resid[((size_t)b * CCH + n) * LL + pos] * scale[n];
                ((float*)outp)[(size_t)m * ldc + n] = v;
            } else {
                v += resid[(size_t)m * ldc + n] * scale[n];
                int b = m / LL, pos = m % LL;
                ((float*)outp)[((size_t)b * CCH + n) * LL + pos] = v;
            }
        }
    }
}

// =================== chunked parallel selective scan ===================
// Recurrence h_l = a_l*h_{l-1} + q_l with a_l = exp(delta*A_n), q_l = delta*B_l*u_l.
// Phase 1: per chunk, compute transfer (P = prod a, S = local solution from h=0).
// Phase 2: tiny sequential pass over 64 chunk boundaries per (b,k,d,n).
// Phase 3: re-run chunks in parallel from corrected start states; emit y via atomics.
// Phases 1/3 compute bit-identical exp values -> deterministic.

__device__ __forceinline__ int dir_pos(int k, int l) {
    int lf = (k & 2) ? (LL - 1 - l) : l;                      // k>=2: flipped scan
    return (k & 1) ? ((lf % HH) * WW + (lf / HH)) : lf;       // k odd: transposed scan
}

__global__ __launch_bounds__(192)
void scan_phase1(const float* __restrict__ xdbl, const __bf16* __restrict__ xiact,
                 const float* __restrict__ dtw, const float* __restrict__ dtb,
                 const float* __restrict__ alogs,
                 float* __restrict__ Pbuf, float* __restrict__ Sbuf)
{
    const int c  = blockIdx.x & (NCHUNK - 1);
    const int bk = blockIdx.x >> 6;          // 0..7 = (b<<2)|k
    const int b = bk >> 2, k = bk & 3;
    const int d = threadIdx.x;
    __shared__ float sh[2][RRANK + 2 * NSTATE];

    float wdt[RRANK];
#pragma unroll
    for (int r = 0; r < RRANK; ++r) wdt[r] = dtw[((size_t)(k * DI + d)) * RRANK + r];
    const float dbias = dtb[k * DI + d];
    float An[NSTATE];
#pragma unroll
    for (int n = 0; n < NSTATE; ++n) An[n] = -__expf(alogs[((size_t)(k * DI + d)) * NSTATE + n]);
    float P[NSTATE], S[NSTATE];
#pragma unroll
    for (int n = 0; n < NSTATE; ++n) { P[n] = 1.f; S[n] = 0.f; }

    const float* xk = xdbl + (size_t)k * MTOT * NPAD;
    for (int l = c * CHLEN; l < (c + 1) * CHLEN; ++l) {
        int row = b * LL + dir_pos(k, l);
        int buf = l & 1;
        if (d < RRANK + 2 * NSTATE) sh[buf][d] = xk[(size_t)row * NPAD + d];
        __syncthreads();
        float dt = dbias;
#pragma unroll
        for (int r = 0; r < RRANK; ++r) dt += sh[buf][r] * wdt[r];
        float delta = (dt > 20.f) ? dt : log1pf(__expf(dt));
        float du = delta * (float)xiact[(size_t)row * DI + d];
#pragma unroll
        for (int n = 0; n < NSTATE; ++n) {
            float dA = __expf(delta * An[n]);
            P[n] *= dA;
            S[n] = S[n] * dA + du * sh[buf][RRANK + n];
        }
    }
    size_t base = ((size_t)bk * DI + d) * NSTATE * NCHUNK;
#pragma unroll
    for (int n = 0; n < NSTATE; ++n) {
        Pbuf[base + n * NCHUNK + c] = P[n];
        Sbuf[base + n * NCHUNK + c] = S[n];
    }
}

__global__ void scan_phase2(const float* __restrict__ Pbuf, const float* __restrict__ Sbuf,
                            float* __restrict__ Hbuf)
{
    int i = blockIdx.x * blockDim.x + threadIdx.x;   // (bk,d,n) flat
    if (i >= 8 * DI * NSTATE) return;
    size_t base = (size_t)i * NCHUNK;
    float h = 0.f;
    for (int c = 0; c < NCHUNK; ++c) {
        Hbuf[base + c] = h;                          // state at chunk start
        h = Pbuf[base + c] * h + Sbuf[base + c];
    }
}

__global__ __launch_bounds__(192)
void scan_phase3(const float* __restrict__ xdbl, const __bf16* __restrict__ xiact,
                 const float* __restrict__ dtw, const float* __restrict__ dtb,
                 const float* __restrict__ alogs, const float* __restrict__ Hbuf,
                 float* __restrict__ yacc)
{
    const int c  = blockIdx.x & (NCHUNK - 1);
    const int bk = blockIdx.x >> 6;
    const int b = bk >> 2, k = bk & 3;
    const int d = threadIdx.x;
    __shared__ float sh[2][RRANK + 2 * NSTATE];

    float wdt[RRANK];
#pragma unroll
    for (int r = 0; r < RRANK; ++r) wdt[r] = dtw[((size_t)(k * DI + d)) * RRANK + r];
    const float dbias = dtb[k * DI + d];
    float An[NSTATE];
#pragma unroll
    for (int n = 0; n < NSTATE; ++n) An[n] = -__expf(alogs[((size_t)(k * DI + d)) * NSTATE + n]);
    float h[NSTATE];
    size_t base = ((size_t)bk * DI + d) * NSTATE * NCHUNK;
#pragma unroll
    for (int n = 0; n < NSTATE; ++n) h[n] = Hbuf[base + n * NCHUNK + c];

    const float* xk = xdbl + (size_t)k * MTOT * NPAD;
    for (int l = c * CHLEN; l < (c + 1) * CHLEN; ++l) {
        int row = b * LL + dir_pos(k, l);
        int buf = l & 1;
        if (d < RRANK + 2 * NSTATE) sh[buf][d] = xk[(size_t)row * NPAD + d];
        __syncthreads();
        float dt = dbias;
#pragma unroll
        for (int r = 0; r < RRANK; ++r) dt += sh[buf][r] * wdt[r];
        float delta = (dt > 20.f) ? dt : log1pf(__expf(dt));
        float du = delta * (float)xiact[(size_t)row * DI + d];
        float y = 0.f;
#pragma unroll
        for (int n = 0; n < NSTATE; ++n) {
            float dA = __expf(delta * An[n]);
            h[n] = h[n] * dA + du * sh[buf][RRANK + n];
            y += h[n] * sh[buf][RRANK + NSTATE + n];
        }
        atomicAdd(&yacc[(size_t)row * DI + d], y);
    }
}

// ---------------- out_norm LN over 192 + SiLU(z) gate -> bf16 ----------------
__global__ void ln_gate_kernel(const float* __restrict__ y, const float* __restrict__ xz,
                               const float* __restrict__ g, const float* __restrict__ be,
                               __bf16* __restrict__ out) {
    int idx = blockIdx.x * blockDim.x + threadIdx.x;
    if (idx >= MTOT) return;
    const float* yp = y + (size_t)idx * DI;
    float mean = 0.f;
    for (int c = 0; c < DI; ++c) mean += yp[c];
    mean *= (1.0f / DI);
    float var = 0.f;
    for (int c = 0; c < DI; ++c) { float dd = yp[c] - mean; var += dd * dd; }
    var *= (1.0f / DI);
    float inv = rsqrtf(var + 1e-5f);
    const float* zp = xz + (size_t)idx * (2 * DI) + DI;
    for (int c = 0; c < DI; ++c) {
        float zc = zp[c];
        float sil = zc / (1.0f + __expf(-zc));
        out[(size_t)idx * DI + c] = (__bf16)(((yp[c] - mean) * inv * g[c] + be[c]) * sil);
    }
}

// ---------------- FFN LN over 96 -> bf16 ----------------
__global__ void ln_rows96_kernel(const float* __restrict__ src, const float* __restrict__ g,
                                 const float* __restrict__ be, __bf16* __restrict__ out) {
    int idx = blockIdx.x * blockDim.x + threadIdx.x;
    if (idx >= MTOT) return;
    const float* sp = src + (size_t)idx * CCH;
    float mean = 0.f;
    for (int c = 0; c < CCH; ++c) mean += sp[c];
    mean *= (1.0f / CCH);
    float var = 0.f;
    for (int c = 0; c < CCH; ++c) { float dd = sp[c] - mean; var += dd * dd; }
    var *= (1.0f / CCH);
    float inv = rsqrtf(var + 1e-5f);
    for (int c = 0; c < CCH; ++c)
        out[(size_t)idx * CCH + c] = (__bf16)((sp[c] - mean) * inv * g[c] + be[c]);
}

extern "C" void kernel_launch(void* const* d_in, const int* in_sizes, int n_in,
                              void* d_out, int out_size, void* d_ws, size_t ws_size,
                              hipStream_t stream)
{
    (void)in_sizes; (void)n_in; (void)out_size; (void)ws_size;
    const float* x          = (const float*)d_in[0];
    const float* ln_in_g    = (const float*)d_in[1];
    const float* ln_in_b    = (const float*)d_in[2];
    const float* in_proj_w  = (const float*)d_in[3];
    const float* in_proj_b  = (const float*)d_in[4];
    const float* conv_w     = (const float*)d_in[5];
    const float* conv_b     = (const float*)d_in[6];
    const float* x_proj_w   = (const float*)d_in[7];
    const float* dt_w       = (const float*)d_in[8];
    const float* dt_b       = (const float*)d_in[9];
    const float* A_logs     = (const float*)d_in[10];
    const float* Ds         = (const float*)d_in[11];
    const float* out_norm_g = (const float*)d_in[12];
    const float* out_norm_b = (const float*)d_in[13];
    const float* out_proj_w = (const float*)d_in[14];
    const float* out_proj_b = (const float*)d_in[15];
    const float* ln_ffn_g   = (const float*)d_in[16];
    const float* ln_ffn_b   = (const float*)d_in[17];
    const float* fc1_w      = (const float*)d_in[18];
    const float* fc1_b      = (const float*)d_in[19];
    const float* fc2_w      = (const float*)d_in[20];
    const float* fc2_b      = (const float*)d_in[21];
    const float* scale1     = (const float*)d_in[22];
    const float* scale2     = (const float*)d_in[23];

    char* ws = (char*)d_ws;
    size_t off = 0;
    auto alloc = [&](size_t bytes) -> void* {
        void* p = ws + off;
        off = (off + bytes + 255) & ~(size_t)255;
        return p;
    };

    __bf16* wb_in  = (__bf16*)alloc((size_t)384 * 96 * 2);
    __bf16* wb_xp  = (__bf16*)alloc((size_t)4 * NPAD * DI * 2);
    __bf16* wb_out = (__bf16*)alloc((size_t)96 * 192 * 2);
    __bf16* wb_fc1 = (__bf16*)alloc((size_t)192 * 96 * 2);
    __bf16* wb_fc2 = (__bf16*)alloc((size_t)96 * 192 * 2);
    __bf16* xn     = (__bf16*)alloc((size_t)MTOT * 96 * 2);
    float*  xzbuf  = (float*) alloc((size_t)MTOT * 384 * 4);
    __bf16* xiact  = (__bf16*)alloc((size_t)MTOT * DI * 2);
    float*  yacc   = (float*) alloc((size_t)MTOT * DI * 4);
    float*  xdbl   = (float*) alloc((size_t)4 * MTOT * NPAD * 4);
    __bf16* ygate  = (__bf16*)alloc((size_t)MTOT * DI * 2);
    float*  x2     = (float*) alloc((size_t)MTOT * 96 * 4);
    __bf16* h1     = (__bf16*)alloc((size_t)MTOT * 96 * 2);
    __bf16* h2     = (__bf16*)alloc((size_t)MTOT * DI * 2);
    size_t scan_elems = (size_t)8 * DI * NSTATE * NCHUNK;     // 1.57M
    float*  Pbuf   = (float*) alloc(scan_elems * 4);
    float*  Sbuf   = (float*) alloc(scan_elems * 4);
    float*  Hbuf   = (float*) alloc(scan_elems * 4);

    // weight staging (bf16, padded where needed)
    cvt_bf16_kernel<<<(36864 + 255) / 256, 256, 0, stream>>>(wb_in, in_proj_w, 36864);
    cvt_bf16_kernel<<<(18432 + 255) / 256, 256, 0, stream>>>(wb_out, out_proj_w, 18432);
    cvt_bf16_kernel<<<(18432 + 255) / 256, 256, 0, stream>>>(wb_fc1, fc1_w, 18432);
    cvt_bf16_kernel<<<(18432 + 255) / 256, 256, 0, stream>>>(wb_fc2, fc2_w, 18432);
    cvt_xproj_kernel<<<(4 * NPAD * DI + 255) / 256, 256, 0, stream>>>(wb_xp, x_proj_w);

    // 1) LN + in_proj GEMM (M=18432, K=96, N=384)
    ln_in_kernel<<<MTOT / 256, 256, 0, stream>>>(x, ln_in_g, ln_in_b, xn);
    gemm_wmma_kernel<0, 4><<<dim3(288, 6, 1), 128, 0, stream>>>(
        xn, 96, wb_in, 0, in_proj_b, xzbuf, 384, 0, 96, nullptr, nullptr);

    // 2) depthwise conv + SiLU (also seeds y accumulator with sum_k Ds * u)
    conv_silu_kernel<<<(MTOT * DI) / 256, 256, 0, stream>>>(xzbuf, conv_w, conv_b, Ds, xiact, yacc);

    // 3) x_proj for all 4 directions at spatial order (M=18432, K=192, N=48pad, z=4)
    gemm_wmma_kernel<0, 3><<<dim3(288, 1, 4), 128, 0, stream>>>(
        xiact, 192, wb_xp, (long long)NPAD * DI, nullptr,
        xdbl, NPAD, (long long)MTOT * NPAD, 192, nullptr, nullptr);

    // 4) chunked parallel selective scan: 512 blocks in phases 1/3
    scan_phase1<<<8 * NCHUNK, 192, 0, stream>>>(xdbl, xiact, dt_w, dt_b, A_logs, Pbuf, Sbuf);
    scan_phase2<<<(8 * DI * NSTATE + 255) / 256, 256, 0, stream>>>(Pbuf, Sbuf, Hbuf);
    scan_phase3<<<8 * NCHUNK, 192, 0, stream>>>(xdbl, xiact, dt_w, dt_b, A_logs, Hbuf, yacc);

    // 5) out_norm + gate, out_proj GEMM + residual*scale1 -> x2
    ln_gate_kernel<<<MTOT / 256, 256, 0, stream>>>(yacc, xzbuf, out_norm_g, out_norm_b, ygate);
    gemm_wmma_kernel<2, 3><<<dim3(288, 2, 1), 128, 0, stream>>>(
        ygate, 192, wb_out, 0, out_proj_b, x2, 96, 0, 192, x, scale1);

    // 6) FFN: LN -> fc1 (+leaky) -> fc2 (+residual*scale2) -> NCHW output
    ln_rows96_kernel<<<MTOT / 256, 256, 0, stream>>>(x2, ln_ffn_g, ln_ffn_b, h1);
    gemm_wmma_kernel<1, 4><<<dim3(288, 3, 1), 128, 0, stream>>>(
        h1, 96, wb_fc1, 0, fc1_b, h2, 192, 0, 96, nullptr, nullptr);
    gemm_wmma_kernel<3, 3><<<dim3(288, 2, 1), 128, 0, stream>>>(
        h2, 192, wb_fc2, 0, fc2_b, d_out, 96, 0, 192, x2, scale2);
}
